// SE3Transformer_31825707663886
// MI455X (gfx1250) — matrix operand order repaired and checked by reading, not compiled, not run on hardware
//
#include <hip/hip_runtime.h>
#include <math.h>

// ---------------------------------------------------------------------------
// CDNA5 (gfx1250) SE(3)-Transformer forward.
// Radial MLPs (the FLOP-dominant part) run on v_wmma_f32_16x16x32_bf16:
//   GEMM1: (16 edges x K=32 padded ef) @ (32 x 32)  -> 2 WMMA
//   GEMM2: (16 edges x K=32 hidden)    @ (32 x OUT) -> OUT/16 WMMA (unrolled)
// Weights are pre-packed once per call into B-fragment-layout bf16 so the
// conv inner loop issues one 32B vector load per WMMA.
// Wave-private LDS staging is ordered with s_wait_dscnt (no block barriers).
// LDS rows use stride = multiple-of-4 floats with stride%64 in {4,20,36}:
// 16B-aligned rows (b128 DS ops) and conflict-free across 64 banks.
// ---------------------------------------------------------------------------

typedef __bf16 bf16_t;
typedef __attribute__((ext_vector_type(16))) __bf16 bf16x16;
typedef __attribute__((ext_vector_type(8)))  float  f32x8;

__device__ inline f32x8 wmma_bf16_16x16x32(bf16x16 a, bf16x16 b, f32x8 c) {
  // 8 args: (neg_a, A, neg_b, B, c_mod, C, reuse_a, reuse_b)
  return __builtin_amdgcn_wmma_f32_16x16x32_bf16(false, a, false, b, (short)0, c,
                                                 false, false);
}

// wave-local LDS ordering: DS ops are in-order per wave; wait for completion
// so cross-lane LDS reads observe prior stores. (CDNA5 split wait counters.)
__device__ inline void lds_fence() {
  asm volatile("s_wait_dscnt 0x0" ::: "memory");
}

// order-preserving float -> uint key for atomicMax-based segment max
__device__ inline unsigned fkey(float x) {
  unsigned u = __float_as_uint(x);
  return (u & 0x80000000u) ? ~u : (u | 0x80000000u);
}
__device__ inline float funkey(unsigned k) {
  unsigned u = (k & 0x80000000u) ? (k & 0x7fffffffu) : ~k;
  return __uint_as_float(u);
}

// ---------------------------------------------------------------------------
// geometry: r, rhat per edge
// ---------------------------------------------------------------------------
__global__ void geom_kernel(const float* __restrict__ pos,
                            const int* __restrict__ ei,
                            float* __restrict__ geom, int E) {
  int e = blockIdx.x * blockDim.x + threadIdx.x;
  if (e >= E) return;
  int r = ei[e], c = ei[E + e];
  float dx = pos[3 * r + 0] - pos[3 * c + 0];
  float dy = pos[3 * r + 1] - pos[3 * c + 1];
  float dz = pos[3 * r + 2] - pos[3 * c + 2];
  float rr = sqrtf(dx * dx + dy * dy + dz * dz + 1e-12f);
  float inv = 1.0f / rr;
  float4 g = make_float4(rr, dx * inv, dy * inv, dz * inv);
  ((float4*)geom)[e] = g;
}

// ---------------------------------------------------------------------------
// Pre-pack radial weights into WMMA B-fragment layout (bf16).
// Fragment entry (tile, lane) holds 16 bf16: element j <-> K=(lane>>4)*16+j,
// column N = tile*16 + (lane&15).  Tiles 0..1: padded W1 (K=5 real rows),
// tiles 2..2+OUT/16-1: W2.
// ---------------------------------------------------------------------------
struct PairPrep { const float* w1; const float* w2; int OUT; int off; };
struct PrepArgs { PairPrep p[32]; int np; };

__global__ void prep_frags_kernel(PrepArgs args, bf16x16* __restrict__ frag) {
  int pi = blockIdx.y;
  if (pi >= args.np) return;
  PairPrep pp = args.p[pi];
  int idx = blockIdx.x * blockDim.x + threadIdx.x;
  int total = (2 + pp.OUT / 16) * 32;
  if (idx >= total) return;
  int lane = idx & 31;
  int t = idx >> 5;
  int ehalf = lane >> 4, l16 = lane & 15, kb = ehalf * 16;
  bf16x16 v;
  if (t < 2) {  // W1 tile (K padded 5 -> 32)
    int Nc = t * 16 + l16;
#pragma unroll
    for (int j = 0; j < 16; ++j) {
      int K = kb + j;
      v[j] = (K < 5) ? (bf16_t)pp.w1[K * 32 + Nc] : (bf16_t)0.f;
    }
  } else {  // W2 tile
    int Nc = (t - 2) * 16 + l16;
#pragma unroll
    for (int j = 0; j < 16; ++j)
      v[j] = (bf16_t)pp.w2[(long)(kb + j) * pp.OUT + Nc];
  }
  frag[pp.off + idx] = v;
}

// ---------------------------------------------------------------------------
// Fused radial-MLP (WMMA) + equivariant contraction per (k,l) pair.
// One wave = one 16-edge tile; 4 independent waves / block.
// ---------------------------------------------------------------------------
template <int KIN, int LOUT, int CI, int CO, int NF, bool ACCUM, bool ATOM>
__global__ __launch_bounds__(128) void edge_conv_kernel(
    const int* __restrict__ ei, const float* __restrict__ efeat,
    const float* __restrict__ geom, const float* __restrict__ hin,
    const bf16x16* __restrict__ frag, const float* __restrict__ b1,
    const float* __restrict__ b2, float* __restrict__ outp, int E) {
  constexpr int MD  = 2 * LOUT + 1;
  constexpr int OUT = NF * CO * CI;
  constexpr int NT  = OUT / 16;
  constexpr int NW  = 4;
  constexpr int TROW = (KIN == 0) ? CI : NF * CI * MD;

  __shared__ float s_h[NW][16][36];        // relu(hmid) 16x32, 16B rows
  __shared__ float s_r[NW][16][20];        // R tile 16x16, 16B rows
  __shared__ float s_t[NW][16][TROW + 4];  // t-tensor / features, 16B rows

  const int lane  = threadIdx.x & 31;
  const int wv    = threadIdx.x >> 5;
  const int ehalf = lane >> 4;  // 0: K-low half, 1: K-high half
  const int l16   = lane & 15;
  const long tile = (long)blockIdx.x * NW + wv;
  const long er   = tile * 16 + l16;  // edge mirrored by this lane
  const bool ev   = (er < E);

  int rowN = 0, colN = 0;
  float rr = 0.f, rh0 = 0.f, rh1 = 0.f, rh2 = 0.f;
  float ef0 = 0.f, ef1 = 0.f, ef2 = 0.f, ef3 = 0.f;
  if (ev) {
    rowN = ei[er];
    colN = ei[E + er];
    float4 g = ((const float4*)geom)[er];
    rr = g.x; rh0 = g.y; rh1 = g.z; rh2 = g.w;
    float4 f = ((const float4*)efeat)[er];
    ef0 = f.x; ef1 = f.y; ef2 = f.z; ef3 = f.w;
  }

  // ---- A1 fragment: 16x32 bf16 of padded ef = [r, edge_feat(4), 0...] ----
  bf16x16 a1;
#pragma unroll
  for (int j = 0; j < 16; ++j) a1[j] = (bf16_t)0.f;
  if (ehalf == 0) {  // only K=0..4 nonzero (zeros if !ev)
    a1[0] = (bf16_t)rr;
    a1[1] = (bf16_t)ef0;
    a1[2] = (bf16_t)ef1;
    a1[3] = (bf16_t)ef2;
    a1[4] = (bf16_t)ef3;
  }

  // ---- GEMM1: hmid = relu(efp @ W1 + b1)  (two 16-wide N tiles) ----
#pragma unroll
  for (int nt = 0; nt < 2; ++nt) {
    bf16x16 bfr = frag[nt * 32 + lane];
    const int Nc = nt * 16 + l16;
    f32x8 c;
#pragma unroll
    for (int i = 0; i < 8; ++i) c[i] = 0.f;
    c = wmma_bf16_16x16x32(a1, bfr, c);
    float bias = b1[Nc];
#pragma unroll
    for (int r = 0; r < 8; ++r) {
      float v = c[r] + bias;
      s_h[wv][r + ehalf * 8][Nc] = (v > 0.f) ? v : 0.f;
    }
  }

  // ---- t-tensor  t[j][i][m'] = sum_n B[m',n,j] f[i,n]  (f = h[KIN][col]) ----
  // No per-element validity select: invalid lanes gather node 0 (safe), and
  // the final write-out is guarded by `ev`.
  if (lane < 16) {
    const float4* f4 = (const float4*)(hin + (long)colN * CI * (2 * KIN + 1));
    if (KIN == 0) {  // store raw f; rhat applied after the i-reduction
#pragma unroll
      for (int i4 = 0; i4 < CI / 4; ++i4) {
        float4 v = f4[i4];
        float* tr = &s_t[wv][lane][4 * i4];
        tr[0] = v.x; tr[1] = v.y; tr[2] = v.z; tr[3] = v.w;
      }
    } else {
#pragma unroll
      for (int i0 = 0; i0 < CI; i0 += 4) {  // 4 features = 12 floats = 3xb128
        float4 va = f4[(i0 * 3) / 4 + 0];
        float4 vb = f4[(i0 * 3) / 4 + 1];
        float4 vc = f4[(i0 * 3) / 4 + 2];
        float fl[12] = {va.x, va.y, va.z, va.w, vb.x, vb.y,
                        vb.z, vb.w, vc.x, vc.y, vc.z, vc.w};
#pragma unroll
        for (int u = 0; u < 4; ++u) {
          const int i = i0 + u;
          float f0 = fl[3 * u + 0], f1 = fl[3 * u + 1], f2 = fl[3 * u + 2];
          if (LOUT == 0) {  // ('1','0'): B = rhat (row vector)
            s_t[wv][lane][i] = rh0 * f0 + rh1 * f1 + rh2 * f2;
          } else {  // ('1','1'): J0 = I, J1 = eps.rhat, J2 = rhat rhat^T - I/3
            float* tr = &s_t[wv][lane][0];
            tr[(0 * CI + i) * 3 + 0] = f0;
            tr[(0 * CI + i) * 3 + 1] = f1;
            tr[(0 * CI + i) * 3 + 2] = f2;
            tr[(1 * CI + i) * 3 + 0] = rh2 * f1 - rh1 * f2;
            tr[(1 * CI + i) * 3 + 1] = rh0 * f2 - rh2 * f0;
            tr[(1 * CI + i) * 3 + 2] = rh1 * f0 - rh0 * f1;
            float d = rh0 * f0 + rh1 * f1 + rh2 * f2;
            tr[(2 * CI + i) * 3 + 0] = rh0 * d - f0 * (1.f / 3.f);
            tr[(2 * CI + i) * 3 + 1] = rh1 * d - f1 * (1.f / 3.f);
            tr[(2 * CI + i) * 3 + 2] = rh2 * d - f2 * (1.f / 3.f);
          }
        }
      }
    }
  }
  lds_fence();  // hmid + t-tensor visible to all lanes of this wave

  // ---- A2 fragment from hmid (16x32, K=32 exactly) ----
  bf16x16 a2;
  {
    const int kb8 = ehalf * 16;
#pragma unroll
    for (int j = 0; j < 8; ++j) a2[j] = (bf16_t)s_h[wv][l16][kb8 + j];
#pragma unroll
    for (int j = 0; j < 8; ++j) a2[8 + j] = (bf16_t)s_h[wv][l16][kb8 + 8 + j];
  }

  float macc[CO * MD];
#pragma unroll
  for (int i = 0; i < CO * MD; ++i) macc[i] = 0.f;

  // ---- GEMM2 tiles + contraction: R = hmid @ W2 + b2; m += R . t ----
#pragma unroll
  for (int t = 0; t < NT; ++t) {
    bf16x16 bfr = frag[(2 + t) * 32 + lane];
    f32x8 c;
#pragma unroll
    for (int i = 0; i < 8; ++i) c[i] = 0.f;
    c = wmma_bf16_16x16x32(a2, bfr, c);
    float bias = b2[t * 16 + l16];
#pragma unroll
    for (int r = 0; r < 8; ++r) s_r[wv][r + ehalf * 8][l16] = c[r] + bias;
    lds_fence();  // R tile visible within wave (DS in-order handles WAR)

    if (lane < 16) {
      // load the R row with b128 DS ops (rows are 16B aligned)
      float rv[16];
      {
        const float4* rrow = (const float4*)&s_r[wv][lane][0];
#pragma unroll
        for (int q = 0; q < 4; ++q) {
          float4 v = rrow[q];
          rv[4 * q + 0] = v.x; rv[4 * q + 1] = v.y;
          rv[4 * q + 2] = v.z; rv[4 * q + 3] = v.w;
        }
      }
      if (KIN == 0) {
        const int o  = t / (CI / 16);
        const int ib = (t % (CI / 16)) * 16;
        const float4* trow = (const float4*)&s_t[wv][lane][ib];
        float s = 0.f;
#pragma unroll
        for (int q = 0; q < 4; ++q) {
          float4 v = trow[q];
          s += rv[4 * q + 0] * v.x + rv[4 * q + 1] * v.y +
               rv[4 * q + 2] * v.z + rv[4 * q + 3] * v.w;
        }
        if (LOUT == 0) {
          macc[o] += s;
        } else {
          macc[o * 3 + 0] += s * rh0;
          macc[o * 3 + 1] += s * rh1;
          macc[o * 3 + 2] += s * rh2;
        }
      } else {
        const int j = t / CO, o = t % CO;
#pragma unroll
        for (int ii = 0; ii < 16; ++ii) {
#pragma unroll
          for (int m = 0; m < MD; ++m)
            macc[o * MD + m] += rv[ii] * s_t[wv][lane][(j * CI + ii) * MD + m];
        }
      }
    }
    lds_fence();  // reads done before next tile overwrites s_r
  }

  if (lane < 16 && ev) {
    if constexpr (ATOM) {  // final conv: segment_sum into out[row]
#pragma unroll
      for (int o = 0; o < CO; ++o)
        atomicAdd(&outp[(long)rowN * CO + o], macc[o]);
    } else {  // per-edge message buffer (CO*MD multiple of 4, 16B aligned)
      float4* d4 = (float4*)(outp + er * (CO * MD));
#pragma unroll
      for (int q = 0; q < (CO * MD) / 4; ++q) {
        float4 v = make_float4(macc[4 * q + 0], macc[4 * q + 1],
                               macc[4 * q + 2], macc[4 * q + 3]);
        if (ACCUM) {
          float4 p = d4[q];
          v.x += p.x; v.y += p.y; v.z += p.z; v.w += p.w;
        }
        d4[q] = v;
      }
    }
  }
}

// ---------------------------------------------------------------------------
// attention + node kernels (small, VALU)
// ---------------------------------------------------------------------------
__global__ void qproj_kernel(const float* __restrict__ h0,
                             const float* __restrict__ h1,
                             const float* __restrict__ q0,
                             const float* __restrict__ q1,
                             float* __restrict__ qbuf, int N, int ci0, int has1) {
  int n = blockIdx.x * blockDim.x + threadIdx.x;
  if (n >= N) return;
  float* qb = qbuf + (long)n * 16;
  for (int o = 0; o < 4; ++o) {
    float s = 0.f;
    for (int i = 0; i < ci0; ++i) s += q0[o * ci0 + i] * h0[(long)n * ci0 + i];
    qb[o] = s;
  }
  if (has1) {
    for (int o = 0; o < 4; ++o)
      for (int m = 0; m < 3; ++m) {
        float s = 0.f;
        for (int i = 0; i < 16; ++i)
          s += q1[o * 16 + i] * h1[((long)n * 16 + i) * 3 + m];
        qb[4 + o * 3 + m] = s;
      }
  }
}

__global__ void logits_kernel(const int* __restrict__ ei,
                              const float* __restrict__ qbuf,
                              const float* __restrict__ k0m,
                              const float* __restrict__ k1m,
                              float* __restrict__ logits,
                              unsigned* __restrict__ mx, int E, int has1,
                              float scale) {
  int e = blockIdx.x * blockDim.x + threadIdx.x;
  if (e >= E) return;
  int row = ei[e];
  const float* qb = qbuf + (long)row * 16;
  const float* k0 = k0m + (long)e * 4;
  float l0 = qb[0] * k0[0] + qb[1] * k0[1];
  float l1 = qb[2] * k0[2] + qb[3] * k0[3];
  if (has1) {
    const float* k1 = k1m + (long)e * 12;
    for (int i = 0; i < 6; ++i) l0 += qb[4 + i] * k1[i];
    for (int i = 6; i < 12; ++i) l1 += qb[4 + i] * k1[i];
  }
  l0 *= scale;
  l1 *= scale;
  logits[2 * e + 0] = l0;
  logits[2 * e + 1] = l1;
  atomicMax(&mx[2 * row + 0], fkey(l0));
  atomicMax(&mx[2 * row + 1], fkey(l1));
}

__global__ void expsum_kernel(const int* __restrict__ ei,
                              float* __restrict__ logits,
                              const unsigned* __restrict__ mx,
                              float* __restrict__ den, int E) {
  int e = blockIdx.x * blockDim.x + threadIdx.x;
  if (e >= E) return;
  int row = ei[e];
  for (int h = 0; h < 2; ++h) {
    float m  = funkey(mx[2 * row + h]);
    float ex = expf(logits[2 * e + h] - m);
    logits[2 * e + h] = ex;
    atomicAdd(&den[2 * row + h], ex);
  }
}

__global__ void agg_kernel(const int* __restrict__ ei,
                           const float* __restrict__ exl,
                           const float* __restrict__ den,
                           const float* __restrict__ v0,
                           const float* __restrict__ v1,
                           float* __restrict__ agg0, float* __restrict__ agg1,
                           int E) {
  int e = blockIdx.x * blockDim.x + threadIdx.x;
  if (e >= E) return;
  int row = ei[e];
  float a[2];
  a[0] = exl[2 * e + 0] / (den[2 * row + 0] + 1e-9f);
  a[1] = exl[2 * e + 1] / (den[2 * row + 1] + 1e-9f);
  for (int o = 0; o < 4; ++o)
    atomicAdd(&agg0[(long)row * 4 + o], a[o >> 1] * v0[(long)e * 4 + o]);
  for (int o = 0; o < 4; ++o)
    for (int m = 0; m < 3; ++m)
      atomicAdd(&agg1[((long)row * 4 + o) * 3 + m],
                a[o >> 1] * v1[((long)e * 4 + o) * 3 + m]);
}

template <int MD>
__global__ void node_update_kernel(
    const float* __restrict__ agg, const float* __restrict__ proj,
    const float* __restrict__ skip, const float* __restrict__ hold, int ci,
    const float* __restrict__ nb1, const float* __restrict__ nb2,
    const float* __restrict__ nw1, const float* __restrict__ nw2,
    float* __restrict__ hnew, int N) {
  int n = blockIdx.x * blockDim.x + threadIdx.x;
  if (n >= N) return;
  float f[16 * MD];
#pragma unroll
  for (int c = 0; c < 16; ++c)
#pragma unroll
    for (int m = 0; m < MD; ++m) {
      float s = 0.f;
#pragma unroll
      for (int i = 0; i < 4; ++i)
        s += proj[c * 4 + i] * agg[((long)n * 4 + i) * MD + m];
      f[c * MD + m] = s;
    }
  if (skip) {
#pragma unroll
    for (int c = 0; c < 16; ++c)
#pragma unroll
      for (int m = 0; m < MD; ++m) {
        float s = 0.f;
        for (int i = 0; i < ci; ++i)
          s += skip[c * ci + i] * hold[((long)n * ci + i) * MD + m];
        f[c * MD + m] += s;
      }
  }
  float nrm[16];
#pragma unroll
  for (int c = 0; c < 16; ++c) {
    float s = 1e-12f;
#pragma unroll
    for (int m = 0; m < MD; ++m) s += f[c * MD + m] * f[c * MD + m];
    nrm[c] = sqrtf(s);
  }
  float smid[16];
#pragma unroll
  for (int j = 0; j < 16; ++j) {
    float s = nb1[j];
#pragma unroll
    for (int c = 0; c < 16; ++c) s += nrm[c] * nw1[j * 16 + c];
    smid[j] = (s > 0.f) ? s : 0.f;
  }
#pragma unroll
  for (int c = 0; c < 16; ++c) {
    float s = nb2[c];
#pragma unroll
    for (int j = 0; j < 16; ++j) s += smid[j] * nw2[c * 16 + j];
    float g = s / nrm[c];
#pragma unroll
    for (int m = 0; m < MD; ++m)
      hnew[((long)n * 16 + c) * MD + m] = g * f[c * MD + m];
  }
}

__global__ void self_kernel(const float* __restrict__ h0,
                            const float* __restrict__ w,
                            float* __restrict__ out, int N) {
  int n = blockIdx.x * blockDim.x + threadIdx.x;
  if (n >= N) return;
  for (int o = 0; o < 32; ++o) {
    float s = 0.f;
    for (int i = 0; i < 16; ++i) s += w[o * 16 + i] * h0[(long)n * 16 + i];
    out[(long)n * 32 + o] += s;
  }
}

// ---------------------------------------------------------------------------
// host side
// ---------------------------------------------------------------------------
struct Rad {
  const float *b1, *b2, *w1, *w2;
};
struct LayerP {
  Rad k00, k01, k10, k11;
  const float *proj0, *proj1, *q0, *q1, *skip0, *skip1;
  Rad v00, v01, v10, v11;
  const float *n0b1, *n0b2, *n0w1, *n0w2, *n1b1, *n1b2, *n1w1, *n1w2;
};

extern "C" void kernel_launch(void* const* d_in, const int* in_sizes, int n_in,
                              void* d_out, int out_size, void* d_ws,
                              size_t ws_size, hipStream_t stream) {
  const float* x     = (const float*)d_in[0];
  const float* pos   = (const float*)d_in[1];
  const int*   ei    = (const int*)d_in[2];
  const float* efeat = (const float*)d_in[3];
  const int N = in_sizes[1] / 3;
  const int E = in_sizes[3] / 4;

  // params flattened in sorted-key pytree order: 'final' < 'layers';
  // rad/norm dicts: b1 < b2 < w1 < w2; attn: kconv<proj<q<skip<vconv.
  int idx = 4;
  auto nxt  = [&]() { return (const float*)d_in[idx++]; };
  auto nxtR = [&]() {
    Rad r;
    r.b1 = nxt(); r.b2 = nxt(); r.w1 = nxt(); r.w2 = nxt();
    return r;
  };
  Rad fc00 = nxtR();
  Rad fc10 = nxtR();
  const float* fself = nxt();
  LayerP L[4] = {};
  L[0].k00 = nxtR();
  L[0].proj0 = nxt(); L[0].proj1 = nxt();
  L[0].q0 = nxt();    L[0].q1 = nullptr;
  L[0].skip0 = nxt(); L[0].skip1 = nullptr;
  L[0].v00 = nxtR();  L[0].v01 = nxtR();
  L[0].n0b1 = nxt(); L[0].n0b2 = nxt(); L[0].n0w1 = nxt(); L[0].n0w2 = nxt();
  L[0].n1b1 = nxt(); L[0].n1b2 = nxt(); L[0].n1w1 = nxt(); L[0].n1w2 = nxt();
  for (int li = 1; li < 4; ++li) {
    L[li].k00 = nxtR(); L[li].k01 = nxtR(); L[li].k10 = nxtR(); L[li].k11 = nxtR();
    L[li].proj0 = nxt(); L[li].proj1 = nxt();
    L[li].q0 = nxt();    L[li].q1 = nxt();
    L[li].skip0 = nxt(); L[li].skip1 = nxt();
    L[li].v00 = nxtR(); L[li].v01 = nxtR(); L[li].v10 = nxtR(); L[li].v11 = nxtR();
    L[li].n0b1 = nxt(); L[li].n0b2 = nxt(); L[li].n0w1 = nxt(); L[li].n0w2 = nxt();
    L[li].n1b1 = nxt(); L[li].n1b2 = nxt(); L[li].n1w1 = nxt(); L[li].n1w2 = nxt();
  }

  // ---- workspace carve-out (bf16 fragment region first: 32B aligned) ----
  PrepArgs prep;
  prep.np = 0;
  int foff = 0;
  auto reg = [&](const Rad& r, int OUT) -> int {
    int off = foff;
    prep.p[prep.np].w1  = r.w1;
    prep.p[prep.np].w2  = r.w2;
    prep.p[prep.np].OUT = OUT;
    prep.p[prep.np].off = off;
    prep.np++;
    foff += (2 + OUT / 16) * 32;
    return off;
  };
  int f_fc00 = reg(fc00, 512);
  int f_fc10 = reg(fc10, 512);
  int fk00[4], fk01[4], fk10[4], fk11[4];
  int fv00[4], fv01[4], fv10[4], fv11[4];
  fk00[0] = reg(L[0].k00, 256);
  fv00[0] = reg(L[0].v00, 256);
  fv01[0] = reg(L[0].v01, 256);
  for (int li = 1; li < 4; ++li) {
    fk00[li] = reg(L[li].k00, 64);
    fk01[li] = reg(L[li].k01, 64);
    fk10[li] = reg(L[li].k10, 64);
    fk11[li] = reg(L[li].k11, 192);
    fv00[li] = reg(L[li].v00, 64);
    fv01[li] = reg(L[li].v01, 64);
    fv10[li] = reg(L[li].v10, 64);
    fv11[li] = reg(L[li].v11, 192);
  }

  bf16x16* fragBase = (bf16x16*)d_ws;
  float* w = (float*)(fragBase + foff);
  auto take = [&](size_t c) { float* p = w; w += c; return p; };
  float* geom    = take((size_t)E * 4);
  float* kmsg0   = take((size_t)E * 4);
  float* kmsg1   = take((size_t)E * 12);
  float* vmsg0   = take((size_t)E * 4);
  float* vmsg1   = take((size_t)E * 12);
  float* logitsb = take((size_t)E * 2);
  float* hA0 = take((size_t)N * 16);
  float* hA1 = take((size_t)N * 48);
  float* hB0 = take((size_t)N * 16);
  float* hB1 = take((size_t)N * 48);
  float* qbuf = take((size_t)N * 16);
  float* nodezero = take((size_t)N * 20);
  // layout: [0,2N) mx keys | [2N,4N) den | [4N,8N) agg0 | [8N,20N) agg1
  unsigned* mx = (unsigned*)nodezero;
  float* den  = nodezero + (size_t)N * 2;
  float* agg0 = nodezero + (size_t)N * 4;
  float* agg1 = nodezero + (size_t)N * 8;

  const int TPB = 256;
  const int nb_e = (E + TPB - 1) / TPB;
  const int nb_n = (N + TPB - 1) / TPB;
  const int nb_c = (E + 63) / 64;  // 4 waves x 16 edges per block

  // pre-pack weight fragments (max total entries per pair: (2+32)*32 = 1088)
  {
    dim3 g((1088 + TPB - 1) / TPB, prep.np);
    prep_frags_kernel<<<g, TPB, 0, stream>>>(prep, fragBase);
  }
  geom_kernel<<<nb_e, TPB, 0, stream>>>(pos, ei, geom, E);

  const float* ins0[4] = {x, hA0, hB0, hA0};
  const float* ins1[4] = {nullptr, hA1, hB1, hA1};
  float* outs0[4] = {hA0, hB0, hA0, hB0};
  float* outs1[4] = {hA1, hB1, hA1, hB1};

  for (int li = 0; li < 4; ++li) {
    const LayerP& P = L[li];
    const float* in0 = ins0[li];
    const float* in1 = ins1[li];
    float* out0 = outs0[li];
    float* out1 = outs1[li];

    if (li == 0) {
      edge_conv_kernel<0, 0, 64, 4, 1, false, false><<<nb_c, 128, 0, stream>>>(
          ei, efeat, geom, in0, fragBase + fk00[0], P.k00.b1, P.k00.b2, kmsg0, E);
      edge_conv_kernel<0, 0, 64, 4, 1, false, false><<<nb_c, 128, 0, stream>>>(
          ei, efeat, geom, in0, fragBase + fv00[0], P.v00.b1, P.v00.b2, vmsg0, E);
      edge_conv_kernel<0, 1, 64, 4, 1, false, false><<<nb_c, 128, 0, stream>>>(
          ei, efeat, geom, in0, fragBase + fv01[0], P.v01.b1, P.v01.b2, vmsg1, E);
    } else {
      edge_conv_kernel<0, 0, 16, 4, 1, false, false><<<nb_c, 128, 0, stream>>>(
          ei, efeat, geom, in0, fragBase + fk00[li], P.k00.b1, P.k00.b2, kmsg0, E);
      edge_conv_kernel<1, 0, 16, 4, 1, true, false><<<nb_c, 128, 0, stream>>>(
          ei, efeat, geom, in1, fragBase + fk10[li], P.k10.b1, P.k10.b2, kmsg0, E);
      edge_conv_kernel<0, 1, 16, 4, 1, false, false><<<nb_c, 128, 0, stream>>>(
          ei, efeat, geom, in0, fragBase + fk01[li], P.k01.b1, P.k01.b2, kmsg1, E);
      edge_conv_kernel<1, 1, 16, 4, 3, true, false><<<nb_c, 128, 0, stream>>>(
          ei, efeat, geom, in1, fragBase + fk11[li], P.k11.b1, P.k11.b2, kmsg1, E);
      edge_conv_kernel<0, 0, 16, 4, 1, false, false><<<nb_c, 128, 0, stream>>>(
          ei, efeat, geom, in0, fragBase + fv00[li], P.v00.b1, P.v00.b2, vmsg0, E);
      edge_conv_kernel<1, 0, 16, 4, 1, true, false><<<nb_c, 128, 0, stream>>>(
          ei, efeat, geom, in1, fragBase + fv10[li], P.v10.b1, P.v10.b2, vmsg0, E);
      edge_conv_kernel<0, 1, 16, 4, 1, false, false><<<nb_c, 128, 0, stream>>>(
          ei, efeat, geom, in0, fragBase + fv01[li], P.v01.b1, P.v01.b2, vmsg1, E);
      edge_conv_kernel<1, 1, 16, 4, 3, true, false><<<nb_c, 128, 0, stream>>>(
          ei, efeat, geom, in1, fragBase + fv11[li], P.v11.b1, P.v11.b2, vmsg1, E);
    }

    (void)hipMemsetAsync(nodezero, 0, (size_t)N * 20 * sizeof(float), stream);

    qproj_kernel<<<nb_n, TPB, 0, stream>>>(in0, in1, P.q0, P.q1, qbuf, N,
                                           li == 0 ? 64 : 16, li == 0 ? 0 : 1);
    logits_kernel<<<nb_e, TPB, 0, stream>>>(ei, qbuf, kmsg0, kmsg1, logitsb, mx,
                                            E, li == 0 ? 0 : 1,
                                            li == 0 ? 0.5f : 0.25f);
    expsum_kernel<<<nb_e, TPB, 0, stream>>>(ei, logitsb, mx, den, E);
    agg_kernel<<<nb_e, TPB, 0, stream>>>(ei, logitsb, den, vmsg0, vmsg1, agg0,
                                         agg1, E);
    node_update_kernel<1><<<nb_n, TPB, 0, stream>>>(
        agg0, P.proj0, P.skip0, in0, li == 0 ? 64 : 16, P.n0b1, P.n0b2, P.n0w1,
        P.n0w2, out0, N);
    node_update_kernel<3><<<nb_n, TPB, 0, stream>>>(
        agg1, P.proj1, li == 0 ? nullptr : P.skip1, li == 0 ? nullptr : in1, 16,
        P.n1b1, P.n1b2, P.n1w1, P.n1w2, out1, N);
  }

  // final: segment_sum of conv messages into out + self term
  float* out = (float*)d_out;
  (void)hipMemsetAsync(out, 0, (size_t)N * 32 * sizeof(float), stream);
  edge_conv_kernel<0, 0, 16, 32, 1, false, true><<<nb_c, 128, 0, stream>>>(
      ei, efeat, geom, hB0, fragBase + f_fc00, fc00.b1, fc00.b2, out, E);
  edge_conv_kernel<1, 0, 16, 32, 1, false, true><<<nb_c, 128, 0, stream>>>(
      ei, efeat, geom, hB1, fragBase + f_fc10, fc10.b1, fc10.b2, out, E);
  self_kernel<<<nb_n, TPB, 0, stream>>>(hB0, fself, out, N);
}
